// CausalAttention_42348377539008
// MI455X (gfx1250) — matrix operand order repaired and checked by reading, not compiled, not run on hardware
//
#include <hip/hip_runtime.h>
#include <hip/hip_bf16.h>

// ---------------------------------------------------------------------------
// CDNA5 (gfx1250) causal attention: QKV GEMM + flash attention + proj GEMM,
// all matrix math through v_wmma_f32_16x16x32_bf16 (wave32).
// ---------------------------------------------------------------------------

typedef __attribute__((ext_vector_type(16))) __bf16 v16bf;
typedef __attribute__((ext_vector_type(8)))  __bf16 v8bf;
typedef __attribute__((ext_vector_type(8)))  float  v8f;

union V16 { v16bf v; v8bf h[2]; };

static __device__ __forceinline__ v8f wmma_bf16(const v16bf& a, const v16bf& b, const v8f& c) {
    return __builtin_amdgcn_wmma_f32_16x16x32_bf16(false, a, false, b, (short)0, c, false, false);
}

static __device__ __forceinline__ v8bf cvt8_bf16(const float4& f0, const float4& f1) {
    v8bf o;
    o[0] = (__bf16)f0.x; o[1] = (__bf16)f0.y; o[2] = (__bf16)f0.z; o[3] = (__bf16)f0.w;
    o[4] = (__bf16)f1.x; o[5] = (__bf16)f1.y; o[6] = (__bf16)f1.z; o[7] = (__bf16)f1.w;
    return o;
}

// ---------------------------------------------------------------------------
// Tiled GEMM: Out[M,N] = A[M,K] (f32) x W[K,N] (f32) + bias, via bf16 WMMA.
// mode 0: QKV epilogue -> scatter Q (scaled by 0.125), K into [H][T][64] and
//         V transposed into [H][64][T], all bf16.
// mode 1: plain f32 output (projection).
// Block: 256 threads (8 waves). Block tile 64x64, K-step 32.
// Wave w: M-subtile (w&3), N-subtiles {2*(w>>2), 2*(w>>2)+1}.
// ---------------------------------------------------------------------------
__global__ __launch_bounds__(256)
void gemm_wmma_kernel(const float* __restrict__ A, const float* __restrict__ W,
                      const float* __restrict__ bias,
                      __bf16* __restrict__ Qb, __bf16* __restrict__ Kb,
                      __bf16* __restrict__ Vt, float* __restrict__ Out,
                      int M, int N, int K, int mode)
{
    __shared__ __bf16 As[64][32];   // [m][k]  bf16
    __shared__ __bf16 Bst[64][32];  // [n][k]  bf16 (B transposed)

    const int tid  = threadIdx.x;
    const int wave = tid >> 5;
    const int lane = tid & 31;
    const int hlf  = lane >> 4;
    const int ml   = lane & 15;

    const int m0 = blockIdx.y * 64;
    const int n0 = blockIdx.x * 64;
    const int mt = wave & 3;
    const int ntBase = (wave >> 2) * 2;

    // Staging index maps (vectorized: one float4-pair per thread per tile).
    const int aRow = tid >> 2, aCol = (tid & 3) * 8;     // A: 64 rows x 32 cols
    const int bK   = tid >> 3, bN   = (tid & 7) * 8;     // W: 32 k-rows x 64 n-cols

    v8f acc[2];
    for (int t = 0; t < 2; ++t)
        for (int r = 0; r < 8; ++r) acc[t][r] = 0.0f;

    for (int k0 = 0; k0 < K; k0 += 32) {
        // Stage A tile (64x32) as bf16: two b128 loads -> one 16B LDS store.
        {
            const float* src = A + (size_t)(m0 + aRow) * K + (k0 + aCol);
            float4 f0 = *(const float4*)(src);
            float4 f1 = *(const float4*)(src + 4);
            *(v8bf*)&As[aRow][aCol] = cvt8_bf16(f0, f1);
        }
        // Stage B tile (32x64) transposed -> Bst[n][k].
        {
            const float* src = W + (size_t)(k0 + bK) * N + (n0 + bN);
            float4 f0 = *(const float4*)(src);
            float4 f1 = *(const float4*)(src + 4);
            v8bf v = cvt8_bf16(f0, f1);
            for (int j = 0; j < 8; ++j) Bst[bN + j][bK] = v[j];
        }
        __syncthreads();

        if (k0 + 32 < K) {  // speculative prefetch of next K-step tiles
            __builtin_prefetch(A + (size_t)(m0 + aRow) * K + (k0 + 32 + aCol), 0, 0);
            __builtin_prefetch(W + (size_t)(k0 + 32 + bK) * N + (n0 + bN), 0, 0);
        }

        // A fragment: lane holds row (mt*16+ml); contiguous K range hlf*16..+15.
        V16 a;
        a.h[0] = *(const v8bf*)&As[mt * 16 + ml][16 * hlf];
        a.h[1] = *(const v8bf*)&As[mt * 16 + ml][16 * hlf + 8];
        for (int t = 0; t < 2; ++t) {
            // B fragment: lane holds column n, contiguous K range hlf*16..+15.
            V16 b;
            b.v = *(const v16bf*)&Bst[(ntBase + t) * 16 + ml][hlf * 16];
            acc[t] = wmma_bf16(a.v, b.v, acc[t]);
        }
        __syncthreads();
    }

    // Epilogue: C layout -> row M = r + 8*hlf, col N = ml.
    for (int t = 0; t < 2; ++t) {
        const int n_g = n0 + (ntBase + t) * 16 + ml;
        const float bv = bias[n_g];
        for (int r = 0; r < 8; ++r) {
            const int m_g = m0 + mt * 16 + r + 8 * hlf;
            const float val = acc[t][r] + bv;
            if (mode == 0) {
                const int sec = n_g / 768;
                const int cc  = n_g % 768;
                const int h   = cc >> 6;
                const int d   = cc & 63;
                if (sec == 0)
                    Qb[((size_t)h * 4096 + m_g) * 64 + d] = (__bf16)(val * 0.125f);
                else if (sec == 1)
                    Kb[((size_t)h * 4096 + m_g) * 64 + d] = (__bf16)val;
                else
                    Vt[((size_t)(h * 64 + d)) * 4096 + m_g] = (__bf16)val;
            } else {
                Out[(size_t)m_g * N + n_g] = val;
            }
        }
    }
}

// ---------------------------------------------------------------------------
// Flash attention. Block = 4 waves = 4 adjacent 16-row query tiles of one
// head. K/V tiles for each 32-key block are staged ONCE in LDS and shared by
// all 4 waves (uniform trip count per block; fully-masked blocks for the
// lower query tiles are benign under -1e30 online-softmax arithmetic).
// Per 32-key block: S = Q*K^T (4 WMMAs), causal mask, online softmax with
// width-16 shuffle reductions, P transposed through per-wave LDS, O += P*V
// (4 WMMAs). Next tile is prefetched (global_prefetch_b8) during compute.
// ---------------------------------------------------------------------------
__global__ __launch_bounds__(128)
void attn_wmma_kernel(const __bf16* __restrict__ Qb, const __bf16* __restrict__ Kb,
                      const __bf16* __restrict__ Vt, float* __restrict__ Ctx)
{
    constexpr int T = 4096, D = 64;
    __shared__ __bf16 Ks[32][64];      // keys kb..kb+31, row-major [key][d]
    __shared__ __bf16 Vts[64][32];     // V d-major [d][key]
    __shared__ __bf16 Ps[4][16][32];   // per-wave P tile [qrow][key]

    const int tid  = threadIdx.x;
    const int wave = tid >> 5;
    const int lane = tid & 31;
    const int hlf  = lane >> 4;
    const int ml   = lane & 15;

    const int h  = blockIdx.y;
    const int qt = blockIdx.x * 4 + wave;   // this wave's 16-row query tile

    // Cooperative-copy index maps (128 threads, 32B each per buffer).
    const int kRow = tid >> 2, kChunk = (tid & 3) * 16;  // Ks: 32 x 64
    const int vRow = tid >> 1, vChunk = (tid & 1) * 16;  // Vts: 64 x 32
    const __bf16* kSrcBase = Kb + ((size_t)h * T + kRow) * D + kChunk;
    const __bf16* vSrcBase = Vt + ((size_t)(h * 64 + vRow)) * T + vChunk;

    // Q fragments for the two 32-wide K-dim steps over D=64.
    V16 aq[2];
    const __bf16* qrow = Qb + ((size_t)h * T + qt * 16 + ml) * D;
    for (int s = 0; s < 2; ++s) {
        aq[s].h[0] = *(const v8bf*)(qrow + s * 32 + 16 * hlf);
        aq[s].h[1] = *(const v8bf*)(qrow + s * 32 + 16 * hlf + 8);
    }

    v8f o[4];
    for (int c = 0; c < 4; ++c)
        for (int r = 0; r < 8; ++r) o[c][r] = 0.0f;
    float mrow[8], lrow[8];
    for (int r = 0; r < 8; ++r) { mrow[r] = -1.0e30f; lrow[r] = 0.0f; }

    const int nIter = 2 * blockIdx.x + 2;   // keys up to block's max query row
    for (int it = 0; it < nIter; ++it) {
        const int kb = it * 32;

        // Stage shared K/V tiles for this key block.
        *(v16bf*)&Ks[kRow][kChunk]  = *(const v16bf*)(kSrcBase + (size_t)kb * D);
        *(v16bf*)&Vts[vRow][vChunk] = *(const v16bf*)(vSrcBase + kb);
        __syncthreads();

        if (it + 1 < nIter) {  // prefetch next key block during compute
            __builtin_prefetch(kSrcBase + (size_t)(kb + 32) * D, 0, 0);
            __builtin_prefetch(vSrcBase + (kb + 32), 0, 0);
        }

        // S tiles for keys [kb..kb+15] and [kb+16..kb+31].
        v8f s0, s1;
        for (int r = 0; r < 8; ++r) { s0[r] = 0.0f; s1[r] = 0.0f; }
        for (int s = 0; s < 2; ++s) {
            v16bf bk0 = *(const v16bf*)&Ks[ml][s * 32 + hlf * 16];
            v16bf bk1 = *(const v16bf*)&Ks[16 + ml][s * 32 + hlf * 16];
            s0 = wmma_bf16(aq[s].v, bk0, s0);
            s1 = wmma_bf16(aq[s].v, bk1, s1);
        }

        // Causal mask + online softmax (row = r + 8*hlf across its 16-lane half).
        float p0[8], p1[8], alpha[8];
        for (int r = 0; r < 8; ++r) {
            const int qpos = qt * 16 + r + 8 * hlf;
            float v0 = (kb + ml      <= qpos) ? s0[r] : -1.0e30f;
            float v1 = (kb + 16 + ml <= qpos) ? s1[r] : -1.0e30f;
            float mx = fmaxf(v0, v1);
            for (int off = 8; off; off >>= 1) mx = fmaxf(mx, __shfl_xor(mx, off, 16));
            const float mnew = fmaxf(mrow[r], mx);
            const float a  = __expf(mrow[r] - mnew);
            const float e0 = __expf(v0 - mnew);
            const float e1 = __expf(v1 - mnew);
            float sum = e0 + e1;
            for (int off = 8; off; off >>= 1) sum += __shfl_xor(sum, off, 16);
            lrow[r] = lrow[r] * a + sum;
            mrow[r] = mnew;
            alpha[r] = a;
            p0[r] = e0; p1[r] = e1;
        }
        for (int c = 0; c < 4; ++c)
            for (int r = 0; r < 8; ++r) o[c][r] *= alpha[r];

        // Transpose P (C layout -> A layout) through this wave's LDS slice.
        for (int r = 0; r < 8; ++r) {
            Ps[wave][r + 8 * hlf][ml]      = (__bf16)p0[r];
            Ps[wave][r + 8 * hlf][16 + ml] = (__bf16)p1[r];
        }
        asm volatile("s_wait_dscnt 0" ::: "memory");   // same-wave LDS RAW across lanes
        V16 ap;
        ap.h[0] = *(const v8bf*)&Ps[wave][ml][16 * hlf];
        ap.h[1] = *(const v8bf*)&Ps[wave][ml][16 * hlf + 8];
        asm volatile("s_wait_dscnt 0" ::: "memory");   // reads done before next overwrite

        // O += P * V from the shared d-major V tile.
        for (int c = 0; c < 4; ++c) {
            v16bf bv = *(const v16bf*)&Vts[c * 16 + ml][hlf * 16];
            o[c] = wmma_bf16(ap.v, bv, o[c]);
        }
        __syncthreads();   // all waves done reading before tiles are restaged
    }

    for (int r = 0; r < 8; ++r) lrow[r] = 1.0f / lrow[r];
    for (int c = 0; c < 4; ++c) {
        for (int r = 0; r < 8; ++r) {
            const int row = qt * 16 + r + 8 * hlf;
            const int col = h * 64 + c * 16 + ml;
            Ctx[(size_t)row * 768 + col] = o[c][r] * lrow[r];
        }
    }
}

// ---------------------------------------------------------------------------
// Launch: qkv gemm -> flash attention -> proj gemm (serialized on stream).
// Workspace: Qb/Kb/Vt bf16 (3 * 6MB) + Ctx f32 (12MB) = 30MB.
// ---------------------------------------------------------------------------
extern "C" void kernel_launch(void* const* d_in, const int* in_sizes, int n_in,
                              void* d_out, int out_size, void* d_ws, size_t ws_size,
                              hipStream_t stream) {
    const float* x      = (const float*)d_in[0];
    const float* W_attn = (const float*)d_in[1];
    const float* b_attn = (const float*)d_in[2];
    const float* W_proj = (const float*)d_in[3];
    const float* b_proj = (const float*)d_in[4];
    float* out = (float*)d_out;

    constexpr int T = 4096, C = 768, H = 12, D = 64;
    const size_t qkvElems = (size_t)H * T * D;          // 3,145,728

    __bf16* Qb = (__bf16*)d_ws;
    __bf16* Kb = Qb + qkvElems;
    __bf16* Vt = Kb + qkvElems;
    float*  Ctx = (float*)(Vt + qkvElems);

    // QKV: [4096 x 768] x [768 x 2304]
    gemm_wmma_kernel<<<dim3((3 * C) / 64, T / 64), dim3(256), 0, stream>>>(
        x, W_attn, b_attn, Qb, Kb, Vt, nullptr, T, 3 * C, C, 0);

    // Flash attention: grid (T/64 query-tile groups, H heads), 4 waves/block.
    attn_wmma_kernel<<<dim3(T / 64, H), dim3(128), 0, stream>>>(Qb, Kb, Vt, Ctx);

    // Projection: [4096 x 768] x [768 x 768]
    gemm_wmma_kernel<<<dim3(C / 64, T / 64), dim3(256), 0, stream>>>(
        Ctx, W_proj, b_proj, nullptr, nullptr, nullptr, out, T, C, C, 1);
}
